// HiddenMarkovModel_53188874993929
// MI455X (gfx1250) — matrix operand order repaired and checked by reading, not compiled, not run on hardware
//
#include <hip/hip_runtime.h>
#include <cstdint>

#define K_TAGS  64
#define V_VOCAB 50000
#define D_DIM   128
#define BATCHN  8192
#define L_LEN   126
#define BOS_TAG 62
#define EOS_TAG 63
#define LOG_EPS -103.2789f   // logf((float)1e-45)

typedef __attribute__((ext_vector_type(16))) _Float16 v16h;
typedef __attribute__((ext_vector_type(8)))  _Float16 v8h;
typedef __attribute__((ext_vector_type(8)))  float    v8f;

static __device__ __forceinline__ v8f wmma_f16(v16h a, v16h b, v8f c) {
  // D = A(16x32 f16) * B(32x16 f16) + C(16x16 f32)
  return __builtin_amdgcn_wmma_f32_16x16x32_f16(false, a, false, b, (short)0, c,
                                                false, false);
}

// ---------------------------------------------------------------------------
// A = softmax(WA, axis=1) with BOS column logit = -inf, then + EPS.
// Store f16 copy (WMMA B-operand source) and f32 copy (final EOS column).
__global__ void k_prepA(const float* __restrict__ WA,
                        float* __restrict__ A32, _Float16* __restrict__ A16) {
  int i = threadIdx.x;
  if (i >= K_TAGS) return;
  float row[K_TAGS];
  float m = -__builtin_inff();
  for (int j = 0; j < K_TAGS; ++j) {
    float x = WA[i * K_TAGS + j];
    if (j == BOS_TAG) x = -__builtin_inff();
    row[j] = x;
    m = fmaxf(m, x);
  }
  float s = 0.f;
  for (int j = 0; j < K_TAGS; ++j) { float e = __expf(row[j] - m); row[j] = e; s += e; }
  float inv = 1.f / s;
  for (int j = 0; j < K_TAGS; ++j) {
    float a = row[j] * inv + 1e-45f;
    A32[i * K_TAGS + j] = a;
    A16[i * K_TAGS + j] = (_Float16)a;
  }
}

// ---------------------------------------------------------------------------
// S[v][k] = dot(ThetaB[k,:], E[v,:])   (64x50000 scores, K-contiguous layout)
// One wave computes a 16(v) x 64(k) tile via 16 f16 WMMAs (D=128 -> 4 k-chunks).
__global__ void __launch_bounds__(128) k_scores(const float* __restrict__ ThetaB,
                                                const float* __restrict__ E,
                                                float* __restrict__ S) {
  int wave = blockIdx.x * 4 + (threadIdx.x >> 5);
  if (wave >= V_VOCAB / 16) return;               // wave-uniform exit
  int lane = threadIdx.x & 31, lr = lane & 15, g = lane >> 4;
  int vbase = wave * 16;

  // A-operand: E rows (16 x 128), f16 16x32 layout per chunk
  const float* Erow = E + (size_t)(vbase + lr) * D_DIM;
  v16h aop[4];
#pragma unroll
  for (int c = 0; c < 4; ++c) {
#pragma unroll
    for (int e = 0; e < 16; ++e) {
      int k = c * 32 + (e < 8 ? 8 * g + e : 16 + 8 * g + (e - 8));
      aop[c][e] = (_Float16)Erow[k];
    }
  }
#pragma unroll
  for (int t = 0; t < 4; ++t) {
    v8f acc = {};
#pragma unroll
    for (int c = 0; c < 4; ++c) {
      // B-operand: ThetaB^T (32 x 16): lane%16 = tag column, element = d-row (table order)
      v16h b;
      int d = c * 32;
#pragma unroll
      for (int e = 0; e < 16; ++e)
        b[e] = (_Float16)ThetaB[(size_t)(t * 16 + lr) * D_DIM + d + (e < 8 ? 8 * g + e : 16 + 8 * g + (e - 8))];
      acc = wmma_f16(aop[c], b, acc);
    }
#pragma unroll
    for (int r = 0; r < 8; ++r)
      S[(size_t)(vbase + r + 8 * g) * K_TAGS + t * 16 + lr] = acc[r];
  }
}

// ---------------------------------------------------------------------------
// Stage-1 streaming logsumexp over V per tag k (online max/sum).
__global__ void k_red1(const float* __restrict__ S,
                       float* __restrict__ pmax, float* __restrict__ psum) {
  __shared__ float sm[256], ss[256];
  int tid = threadIdx.x;
  int k = tid & 63, vg = tid >> 6;
  int per = (V_VOCAB + gridDim.x - 1) / gridDim.x;
  int v0 = blockIdx.x * per;
  int v1 = v0 + per; if (v1 > V_VOCAB) v1 = V_VOCAB;
  float M = -__builtin_inff(), Sm = 0.f;
  for (int v = v0 + vg; v < v1; v += 4) {
    float x = S[(size_t)v * K_TAGS + k];
    if (x > M) { Sm = Sm * __expf(M - x) + 1.f; M = x; }
    else       { Sm += __expf(x - M); }
  }
  sm[tid] = M; ss[tid] = Sm;
  __syncthreads();
  if (vg == 0) {
#pragma unroll
    for (int j = 1; j < 4; ++j) {
      float M2 = sm[tid + 64 * j], S2 = ss[tid + 64 * j];
      if (M2 > M) { Sm = Sm * __expf(M - M2) + S2; M = M2; }
      else        { Sm += S2 * __expf(M2 - M); }
    }
    pmax[blockIdx.x * 64 + k] = M;
    psum[blockIdx.x * 64 + k] = Sm;
  }
}

__global__ void k_red2(const float* __restrict__ pmax, const float* __restrict__ psum,
                       float* __restrict__ logZB) {
  int k = threadIdx.x;
  if (k >= K_TAGS) return;
  float M = -__builtin_inff(), Sm = 0.f;
  for (int b = 0; b < 128; ++b) {
    float M2 = pmax[b * 64 + k], S2 = psum[b * 64 + k];
    if (M2 > M) { Sm = Sm * __expf(M - M2) + S2; M = M2; }
    else        { Sm += S2 * __expf(M2 - M); }
  }
  logZB[k] = M + __logf(Sm);
}

// In place: S -> logB = S - logZB[k]; BOS/EOS rows -> log(EPS).
__global__ void k_fixB(float* __restrict__ S, const float* __restrict__ logZB) {
  size_t idx = (size_t)blockIdx.x * blockDim.x + threadIdx.x;
  if (idx >= (size_t)V_VOCAB * K_TAGS) return;
  int k = (int)(idx & 63);
  S[idx] = (k >= BOS_TAG) ? LOG_EPS : S[idx] - logZB[k];
}

// ---------------------------------------------------------------------------
// Forward recursion, linear space + per-step rescale.
// One wave owns 16 sentences; alpha tile p[16 x 64] lives in 4 WMMA C accums.
__global__ void __launch_bounds__(128) k_forward(const int* __restrict__ words,
                                                 const float* __restrict__ logB,
                                                 const _Float16* __restrict__ A16,
                                                 const float* __restrict__ A32,
                                                 float* __restrict__ out) {
  __shared__ _Float16 lds[4 * 16 * K_TAGS];       // per-wave 16x64 f16 staging
  int wslot = (threadIdx.x >> 5) * 16 * K_TAGS;
  int wave = blockIdx.x * 4 + (threadIdx.x >> 5); // 512 waves == BATCH/16 exactly
  int lane = threadIdx.x & 31, lr = lane & 15, g = lane >> 4;
  int sbase = wave * 16;

  // Preload transition matrix as WMMA B-operands: Af[c][t], lane%16 = column j, element = i (table order)
  v16h Af[2][4];
#pragma unroll
  for (int c = 0; c < 2; ++c)
#pragma unroll
    for (int t = 0; t < 4; ++t)
#pragma unroll
      for (int e = 0; e < 16; ++e)
        Af[c][t][e] = A16[(size_t)(c * 32 + (e < 8 ? 8 * g + e : 16 + 8 * g + (e - 8))) * K_TAGS + t * 16 + lr];

  // alpha0: one-hot at BOS (tag 62 -> n-tile 3, column 14), log-scale c = 0
  v8f acc[4];
#pragma unroll
  for (int t = 0; t < 4; ++t) acc[t] = {};
  if (lr == 14) {
#pragma unroll
    for (int r = 0; r < 8; ++r) acc[3][r] = 1.f;
  }
  v8f cvec = {};

  for (int l = 0; l < L_LEN; ++l) {
    // ---- stage p (C layout, f32) -> LDS f16 [m][i] for the A-operand transpose
#pragma unroll
    for (int t = 0; t < 4; ++t)
#pragma unroll
      for (int r = 0; r < 8; ++r)
        lds[wslot + (r + 8 * g) * K_TAGS + t * 16 + lr] = (_Float16)acc[t][r];
    __builtin_amdgcn_wave_barrier();
    asm volatile("s_wait_dscnt 0" ::: "memory");   // DS in-order per wave

    // ---- load p as A-operand fragments (f16 16x32 layout): two b128 DS loads each
    v16h aop[2];
#pragma unroll
    for (int c = 0; c < 2; ++c) {
      const v8h* plo = (const v8h*)&lds[wslot + lr * K_TAGS + c * 32 + 8 * g];
      const v8h* phi = (const v8h*)&lds[wslot + lr * K_TAGS + c * 32 + 16 + 8 * g];
      v8h lo = *plo, hi = *phi;
      aop[c] = __builtin_shufflevector(lo, hi, 0, 1, 2, 3, 4, 5, 6, 7,
                                       8, 9, 10, 11, 12, 13, 14, 15);
    }
    __builtin_amdgcn_wave_barrier();

    // ---- t = p @ A  (8 WMMA)
    v8f nz[4];
#pragma unroll
    for (int t = 0; t < 4; ++t) {
      v8f z = {};
      z = wmma_f16(aop[0], Af[0][t], z);
      z = wmma_f16(aop[1], Af[1][t], z);
      nz[t] = z;
    }

    // ---- emission: t[m,j] *= exp(logB[j, word[m,l]]) ; logB is L2-resident
    int wl = words[(size_t)(sbase + lr) * L_LEN + l];
#pragma unroll
    for (int r = 0; r < 8; ++r) {
      int wr = __shfl(wl, r + 8 * g, 32);          // word of sentence m = r+8g
      const float* bp = logB + (size_t)wr * K_TAGS + lr;
#pragma unroll
      for (int t = 0; t < 4; ++t)
        nz[t][r] *= __expf(bp[t * 16]);
    }

    // ---- rescale: mu = row max; p = t/mu; c += log(mu)
#pragma unroll
    for (int r = 0; r < 8; ++r) {
      float mu = fmaxf(fmaxf(nz[0][r], nz[1][r]), fmaxf(nz[2][r], nz[3][r]));
#pragma unroll
      for (int msk = 1; msk < 16; msk <<= 1)
        mu = fmaxf(mu, __shfl_xor(mu, msk, 32));
      float inv = 1.f / mu;
#pragma unroll
      for (int t = 0; t < 4; ++t) nz[t][r] *= inv;
      cvec[r] += __logf(mu);
    }
#pragma unroll
    for (int t = 0; t < 4; ++t) acc[t] = nz[t];
  }

  // ---- logZ = c + log( sum_i p[m,i] * A[i, EOS] )
  float ae[4];
#pragma unroll
  for (int t = 0; t < 4; ++t)
    ae[t] = A32[(size_t)(t * 16 + lr) * K_TAGS + EOS_TAG];
#pragma unroll
  for (int r = 0; r < 8; ++r) {
    float s = acc[0][r] * ae[0] + acc[1][r] * ae[1] +
              acc[2][r] * ae[2] + acc[3][r] * ae[3];
#pragma unroll
    for (int msk = 1; msk < 16; msk <<= 1)
      s += __shfl_xor(s, msk, 32);
    if (lr == 0) out[sbase + 8 * g + r] = cvec[r] + __logf(s);
  }
}

// ---------------------------------------------------------------------------
extern "C" void kernel_launch(void* const* d_in, const int* in_sizes, int n_in,
                              void* d_out, int out_size, void* d_ws, size_t ws_size,
                              hipStream_t stream) {
  const int*   words  = (const int*)d_in[0];
  const float* ThetaB = (const float*)d_in[1];
  const float* WA     = (const float*)d_in[2];
  const float* E      = (const float*)d_in[3];
  float* out = (float*)d_out;

  char* ws = (char*)d_ws;
  float* S = (float*)ws;                                   // V*K f32 (12.8 MB)
  size_t off = (size_t)V_VOCAB * K_TAGS * sizeof(float);
  _Float16* A16 = (_Float16*)(ws + off); off += (size_t)K_TAGS * K_TAGS * sizeof(_Float16);
  float* A32   = (float*)(ws + off); off += (size_t)K_TAGS * K_TAGS * sizeof(float);
  float* logZB = (float*)(ws + off); off += (size_t)K_TAGS * sizeof(float);
  float* pmax  = (float*)(ws + off); off += 128 * K_TAGS * sizeof(float);
  float* psum  = (float*)(ws + off); off += 128 * K_TAGS * sizeof(float);
  (void)in_sizes; (void)n_in; (void)out_size; (void)ws_size;

  k_prepA<<<1, 64, 0, stream>>>(WA, A32, A16);
  k_scores<<<(V_VOCAB / 16 + 3) / 4, 128, 0, stream>>>(ThetaB, E, S);
  k_red1<<<128, 256, 0, stream>>>(S, pmax, psum);
  k_red2<<<1, 64, 0, stream>>>(pmax, psum, logZB);
  k_fixB<<<((size_t)V_VOCAB * K_TAGS + 255) / 256, 256, 0, stream>>>(S, logZB);
  k_forward<<<BATCHN / 16 / 4, 128, 0, stream>>>(words, S, A16, A32, out);
}
